// AMEncoder_13108240188120
// MI455X (gfx1250) — compile-verified, matrix operand order checked by reading, not run on hardware
//
#include <hip/hip_runtime.h>
#include <hip/hip_bf16.h>

// ============================================================================
// AM_Encoder for MI455X (gfx1250, wave32, WMMA).
//
// Input leaf ordering (setup_inputs() insertion order, params flattened
// recursively in insertion order, _lin = {w, b}):
//  0 h[20000,64] 1 x[20000,4,3] 2 ctx_edges[2,160000] 3 inter_mask[10000]
//  4 inter_x[10000,4,3] 5 inter_edges[2,80000] 6 update_mask[5000]
//  7 inter_update_mask[5000] 8 channel_attr[20000,4,16] 9 channel_weights[20000,4]
// 10 lin_in.w 11 lin_in.b 12 lin_out.w 13 lin_out.b
// 14..26 out_gcl {radial.w,radial.b,edge1.w,edge1.b,edge2.w,edge2.b,
//                 node1.w,node1.b,node2.w,node2.b,coord1.w,coord1.b,coord2.w}
// 27..39 ctx_gcl[0]  40..52 ctx_gcl[1]  53..65 inter_gcl[0]  66..78 inter_gcl[1]
// 79 align_gate.w 80 align_gate.b 81 align1.w 82 align1.b 83 align2.w 84 align2.b
//
// Output: [h(20000*64) | x(20000*12) | inter_x(10000*12)] fp32, flat.
// ============================================================================

#define NN     20000
#define NI_    10000
#define MM_    5000
#define E_CTX  160000
#define E_INT  80000
#define IN_NF  64
#define HID    128
#define OUT_NF 64
#define CC     4
#define CH_NF  16
#define RAD_NF 16

typedef __attribute__((ext_vector_type(16))) _Float16 v16h;
typedef __attribute__((ext_vector_type(8)))  _Float16 v8h;
typedef __attribute__((ext_vector_type(8)))  float    v8f;
typedef unsigned long long u64_t;

// Force global-address-space vector loads (avoid flat_load from generic ptrs).
// Only builtin (trivially copyable) types: scalar u64 and ext-vector halves.
__device__ __forceinline__ u64_t ldg_u64(const _Float16* p) {
  return *(const __attribute__((address_space(1))) u64_t*)p;
}
__device__ __forceinline__ v16h ldg_v16h(const _Float16* p) {
  return *(const __attribute__((address_space(1))) v16h*)p;
}

// A fragment from LDS tile (16x32 f16 row-major), ISA 7.12.2 16-bit A layout.
__device__ __forceinline__ v16h frag_a(const _Float16* As, int lane) {
  int m  = lane & 15;
  int kb = (lane >> 4) * 8;
  v8h lo = *(const v8h*)(As + m * 32 + kb);
  v8h hi = *(const v8h*)(As + m * 32 + 16 + kb);
  v16h a;
#pragma unroll
  for (int i = 0; i < 8; ++i) { a[i] = lo[i]; a[8 + i] = hi[i]; }
  return a;
}

// ---------------------------------------------------------------------------
// Weight pre-pack: fp32 W[K,N] -> f16 fragments in B-lane order.
// Packed element t = ((chunk*nColTiles + colTile)*32 + lane)*16 + j, where
// lane n=lane&15, khalf=(lane>>4)*8; j<8 -> k=khalf+j ; j>=8 -> k=16+khalf+j-8.
// Out-of-range K/N baked as zeros.
// ---------------------------------------------------------------------------
__global__ void k_pack_w(const float* __restrict__ W, _Float16* __restrict__ Wp,
                         int K, int N, int nColTiles, int nChunks) {
  long total = (long)nChunks * nColTiles * 512;
  long t = (long)blockIdx.x * 256 + threadIdx.x;
  if (t >= total) return;
  int  j    = (int)(t & 15);
  int  lane = (int)((t >> 4) & 31);
  long g    = t >> 9;
  long ct   = g % nColTiles;
  long ch   = g / nColTiles;
  int n     = lane & 15;
  int khalf = (lane >> 4) * 8;
  int k     = (j < 8) ? (khalf + j) : (16 + khalf + (j - 8));
  long gk = ch * 32 + k;
  long gn = ct * 16 + n;
  float v = (gk < K && gn < N) ? W[gk * N + gn] : 0.0f;
  Wp[t] = (_Float16)v;
}

// ---------------------------------------------------------------------------
// Fused-concat WMMA GEMM:  Y[M,N] = act([S0(i0)|S1(i1)|S2(i2)] @ W + bias)
// Block: 256 threads = 8 waves; block owns 16 rows x 128 cols; wave w owns
// 16 cols. Threads 0..127 stage the 16x32 f16 A chunk (4 halves each, 8 threads per row, one
// b64 global load + ds_store_b64); all waves run one WMMA per 32-K chunk.
// Segment K widths must be multiples of 4; segment starts are multiples of 32.
// ---------------------------------------------------------------------------
struct GemmArgs {
  const _Float16* s0; const _Float16* s1; const _Float16* s2;
  const int* i0; const int* i1; const int* i2;
  int ld0, ld1, ld2;
  int k0, k1, k2;
  const _Float16* Wp;    // packed fragments
  int  nct;              // nColTiles used when packing
  const float* bias;     // nullptr allowed
  float*      outF;      // nullptr allowed
  _Float16*   outH;      // nullptr allowed
  int  ldo;
  long M; int N;
  int  act;              // 0 none, 1 silu, 2 sigmoid
  int  accum;            // outF += result (residual)
};

__global__ __launch_bounds__(256) void k_gemm(GemmArgs g) {
  __shared__ __attribute__((aligned(16))) _Float16 As[16 * 32];

  const long m0   = (long)blockIdx.x * 16;
  const int  tid  = threadIdx.x;
  const int  lane = tid & 31;
  const int  wave = tid >> 5;
  const int  colTile = blockIdx.y * 8 + wave;
  const bool waveActive = (colTile * 16 < g.N);   // wave-uniform -> EXEC full

  const bool stager = (tid < 128);
  const int  sm = tid >> 3;                       // staging row 0..15
  const int  sk = (tid & 7) * 4;                  // staging K offset {0,4,..,28}

  const _Float16* sp[3] = { g.s0, g.s1, g.s2 };
  const int* ip[3] = { g.i0, g.i1, g.i2 };
  const int ldv[3] = { g.ld0, g.ld1, g.ld2 };
  const int kwv[3] = { g.k0,  g.k1,  g.k2  };

  v8f acc = {};
  int kstart = 0;
  for (int seg = 0; seg < 3; ++seg) {
    const int kw = kwv[seg];
    if (kw <= 0) continue;
    const _Float16* rowp = nullptr;
    if (stager) {
      const int* idx = ip[seg];
      long r = idx ? (long)idx[m0 + sm] : (m0 + sm);
      rowp = sp[seg] + r * (long)ldv[seg];
    }
    for (int kb = 0; kb < kw; kb += 32) {
      __syncthreads();
      if (stager) {
        int kk = kb + sk;
        u64_t v = 0;
        if (kk < kw) v = ldg_u64(rowp + kk);      // kw % 4 == 0 always
        *(u64_t*)(As + sm * 32 + sk) = v;
      }
      __syncthreads();
      if (waveActive) {
        v16h a = frag_a(As, lane);
        long chunk = (long)(kstart + kb) >> 5;
        v16h b = ldg_v16h(g.Wp + ((chunk * g.nct + colTile) * 32 + lane) * 16);
        acc = __builtin_amdgcn_wmma_f32_16x16x32_f16(
                  false, a, false, b, (short)0, acc, false, false);
      }
    }
    kstart += kw;
  }

  if (waveActive) {
    int n  = lane & 15;
    int mb = (lane >> 4) * 8;
    int col = colTile * 16 + n;
    if (col < g.N) {
      float bv = g.bias ? g.bias[col] : 0.0f;
#pragma unroll
      for (int r = 0; r < 8; ++r) {
        long row = m0 + mb + r;
        float v = acc[r] + bv;
        if (g.act == 1)       v = v / (1.0f + __expf(-v));     // silu
        else if (g.act == 2)  v = 1.0f / (1.0f + __expf(-v));  // sigmoid
        if (g.outF) {
          if (g.accum) g.outF[row * g.ldo + col] += v;
          else         g.outF[row * g.ldo + col]  = v;
        }
        if (g.outH) g.outH[row * g.ldo + col] = (_Float16)v;
      }
    }
  }
}

// ---------------------------------------------------------------------------
// Elementwise / graph kernels
// ---------------------------------------------------------------------------
__global__ void k_f32_to_f16(_Float16* dst, const float* src, long n) {
  long t = (long)blockIdx.x * 256 + threadIdx.x;
  if (t < n) dst[t] = (_Float16)src[t];
}
__global__ void k_copy_f32(float* dst, const float* src, long n) {
  long t = (long)blockIdx.x * 256 + threadIdx.x;
  if (t < n) dst[t] = src[t];
}
__global__ void k_zero_f32(float* p, long n) {
  long t = (long)blockIdx.x * 256 + threadIdx.x;
  if (t < n) p[t] = 0.0f;
}
__global__ void k_gather_rows(float* dst, const float* src, const int* idx, long rows, int w) {
  long t = (long)blockIdx.x * 256 + threadIdx.x;
  if (t >= rows * w) return;
  long r = t / w; int c = (int)(t - r * w);
  dst[t] = src[(long)idx[r] * w + c];
}
__global__ void k_scatter_rows(float* dst, const float* src, const int* idx, long rows, int w) {
  long t = (long)blockIdx.x * 256 + threadIdx.x;
  if (t >= rows * w) return;
  long r = t / w; int c = (int)(t - r * w);
  dst[(long)idx[r] * w + c] = src[t];
}
__global__ void k_gather_scatter(float* dst, const float* src, const int* didx,
                                 const int* sidx, long rows, int w) {
  long t = (long)blockIdx.x * 256 + threadIdx.x;
  if (t >= rows * w) return;
  long r = t / w; int c = (int)(t - r * w);
  dst[(long)didx[r] * w + c] = src[(long)sidx[r] * w + c];
}

// Radial features: cd = x[row]-x[col]; Gram(4x4)*mask -> radial lin(16->16) -> f16
__global__ __launch_bounds__(128) void k_radial(
    const float* __restrict__ x, const int* __restrict__ row, const int* __restrict__ col,
    const float* __restrict__ chw, const float* __restrict__ Wr, const float* __restrict__ br,
    float* __restrict__ cd, _Float16* __restrict__ rad16, long E) {
  __shared__ float sW[256];
  __shared__ float sB[16];
  for (int i = threadIdx.x; i < 256; i += 128) sW[i] = Wr[i];
  if (threadIdx.x < 16) sB[threadIdx.x] = br[threadIdx.x];
  __syncthreads();
  long e = (long)blockIdx.x * 128 + threadIdx.x;
  if (e >= E) return;
  int r = row[e], c = col[e];
  float d[12];
#pragma unroll
  for (int i = 0; i < 12; ++i) {
    d[i] = x[(long)r * 12 + i] - x[(long)c * 12 + i];
    cd[e * 12 + i] = d[i];
  }
  float wr[4], wc[4];
#pragma unroll
  for (int i = 0; i < 4; ++i) { wr[i] = chw[(long)r * 4 + i]; wc[i] = chw[(long)c * 4 + i]; }
  float g[16];
#pragma unroll
  for (int a = 0; a < 4; ++a)
#pragma unroll
    for (int b = 0; b < 4; ++b) {
      float s = d[a*3]*d[b*3] + d[a*3+1]*d[b*3+1] + d[a*3+2]*d[b*3+2];
      g[a * 4 + b] = s * wr[a] * wc[b];
    }
#pragma unroll
  for (int j = 0; j < 16; ++j) {
    float s = sB[j];
#pragma unroll
    for (int k = 0; k < 16; ++k) s += g[k] * sW[k * 16 + j];
    rad16[e * 16 + j] = (_Float16)s;
  }
}

__global__ void k_msg_agg(const _Float16* __restrict__ m, const int* __restrict__ row,
                          float* __restrict__ agg, long total) {
  long t = (long)blockIdx.x * 256 + threadIdx.x;
  if (t >= total) return;
  int  f = (int)(t & (HID - 1));
  long e = t >> 7;
  atomicAdd(&agg[(long)row[e] * HID + f], (float)m[t]);
}

__global__ void k_coord_agg(const float* __restrict__ cd, const float* __restrict__ cw,
                            const int* __restrict__ row, float* __restrict__ aggx,
                            float* __restrict__ cnt, long E) {
  long e = (long)blockIdx.x * 256 + threadIdx.x;
  if (e >= E) return;
  int r = row[e];
#pragma unroll
  for (int c = 0; c < 4; ++c) {
    float w = cw[e * 4 + c];
#pragma unroll
    for (int i = 0; i < 3; ++i)
      atomicAdd(&aggx[(long)r * 12 + c * 3 + i], cd[e * 12 + c * 3 + i] * w);
  }
  atomicAdd(&cnt[r], 1.0f);
}
__global__ void k_coord_update(float* __restrict__ x, const float* __restrict__ aggx,
                               const float* __restrict__ cnt, long n12) {
  long t = (long)blockIdx.x * 256 + threadIdx.x;
  if (t >= n12) return;
  float c = cnt[t / 12];
  x[t] += aggx[t] / fmaxf(c, 1.0f);
}

__global__ void k_node_ch(const float* __restrict__ cha, const float* __restrict__ chw,
                          const int* __restrict__ idx, _Float16* __restrict__ out, long n) {
  long i = (long)blockIdx.x * 256 + threadIdx.x;
  if (i >= n) return;
  long s = idx ? idx[i] : i;
  float w[4];
#pragma unroll
  for (int c = 0; c < 4; ++c) w[c] = chw[s * 4 + c];
#pragma unroll
  for (int j = 0; j < 16; ++j) {
    float v = 0.0f;
#pragma unroll
    for (int c = 0; c < 4; ++c) v += cha[(s * 4 + c) * 16 + j] * w[c];
    out[i * 16 + j] = (_Float16)v;
  }
}

__global__ void k_align_combine(float* __restrict__ ih, const float* __restrict__ g,
                                const float* __restrict__ a, long n) {
  long t = (long)blockIdx.x * 256 + threadIdx.x;
  if (t >= n) return;
  float gg = g[t];
  ih[t] = gg * ih[t] + (1.0f - gg) * a[t];
}

// ---------------------------------------------------------------------------
// Host orchestration
// ---------------------------------------------------------------------------
static inline long cdivl(long a, long b) { return (a + b - 1) / b; }

static void gemm(hipStream_t st,
                 const _Float16* s0, const int* i0, int ld0, int k0,
                 const _Float16* s1, const int* i1, int ld1, int k1,
                 const _Float16* s2, const int* i2, int ld2, int k2,
                 const _Float16* Wp, const float* bias,
                 float* outF, _Float16* outH, int ldo,
                 long M, int N, int act, int accum) {
  GemmArgs a;
  a.s0 = s0; a.s1 = s1; a.s2 = s2;
  a.i0 = i0; a.i1 = i1; a.i2 = i2;
  a.ld0 = ld0; a.ld1 = ld1; a.ld2 = ld2;
  a.k0 = k0; a.k1 = k1; a.k2 = k2;
  a.Wp = Wp; a.nct = (N + 15) / 16;
  a.bias = bias; a.outF = outF; a.outH = outH; a.ldo = ldo;
  a.M = M; a.N = N; a.act = act; a.accum = accum;
  dim3 grid((unsigned)(M / 16), (unsigned)((N + 127) / 128));
  k_gemm<<<grid, 256, 0, st>>>(a);
}

struct GclW {
  const float *rw, *rb, *e1w, *e1b, *e2w, *e2b, *n1w, *n1b, *n2w, *n2b, *c1w, *c1b, *c2w;
};
static GclW gclw(void* const* d_in, int b) {
  GclW g;
  g.rw  = (const float*)d_in[b+0];  g.rb  = (const float*)d_in[b+1];
  g.e1w = (const float*)d_in[b+2];  g.e1b = (const float*)d_in[b+3];
  g.e2w = (const float*)d_in[b+4];  g.e2b = (const float*)d_in[b+5];
  g.n1w = (const float*)d_in[b+6];  g.n1b = (const float*)d_in[b+7];
  g.n2w = (const float*)d_in[b+8];  g.n2b = (const float*)d_in[b+9];
  g.c1w = (const float*)d_in[b+10]; g.c1b = (const float*)d_in[b+11];
  g.c2w = (const float*)d_in[b+12];
  return g;
}
struct GclP {   // packed f16 weights for one GCL
  const _Float16 *e1, *e2, *n1, *n2, *c1, *c2;
};

struct GclWs {  // scratch shared between ctx/inter GCLs (sized for the larger)
  float* cd; _Float16* rad16; _Float16* m1; _Float16* mm; float* cw;
  float* agg; _Float16* agg16; float* aggx; float* cnt;
};

static void run_gcl(hipStream_t st, float* h, _Float16* h16, float* x,
                    const int* row, const int* col, const float* chw,
                    const _Float16* nch16, const GclW& W, const GclP& P,
                    long n, long E, const GclWs& s) {
  k_radial<<<(unsigned)cdivl(E, 128), 128, 0, st>>>(x, row, col, chw, W.rw, W.rb,
                                                    s.cd, s.rad16, E);
  // edge MLP (gathers fused into GEMM A staging)
  gemm(st, h16, row, HID, HID, h16, col, HID, HID, s.rad16, nullptr, RAD_NF, RAD_NF,
       P.e1, W.e1b, nullptr, s.m1, HID, E, HID, /*silu*/1, 0);
  gemm(st, s.m1, nullptr, HID, HID, nullptr, nullptr, 0, 0, nullptr, nullptr, 0, 0,
       P.e2, W.e2b, nullptr, s.mm, HID, E, HID, /*silu*/1, 0);
  // coord MLP (c1 reuses m1 buffer)
  gemm(st, s.mm, nullptr, HID, HID, nullptr, nullptr, 0, 0, nullptr, nullptr, 0, 0,
       P.c1, W.c1b, nullptr, s.m1, HID, E, HID, /*silu*/1, 0);
  gemm(st, s.m1, nullptr, HID, HID, nullptr, nullptr, 0, 0, nullptr, nullptr, 0, 0,
       P.c2, nullptr, s.cw, nullptr, CC, E, CC, 0, 0);
  // zero accumulators
  k_zero_f32<<<(unsigned)cdivl(n * HID, 256), 256, 0, st>>>(s.agg, n * HID);
  k_zero_f32<<<(unsigned)cdivl(n * 12, 256), 256, 0, st>>>(s.aggx, n * 12);
  k_zero_f32<<<(unsigned)cdivl(n, 256), 256, 0, st>>>(s.cnt, n);
  // segment sums + coord update
  k_msg_agg<<<(unsigned)cdivl(E * HID, 256), 256, 0, st>>>(s.mm, row, s.agg, E * HID);
  k_coord_agg<<<(unsigned)cdivl(E, 256), 256, 0, st>>>(s.cd, s.cw, row, s.aggx, s.cnt, E);
  k_coord_update<<<(unsigned)cdivl(n * 12, 256), 256, 0, st>>>(x, s.aggx, s.cnt, n * 12);
  // node MLP with residual
  k_f32_to_f16<<<(unsigned)cdivl(n * HID, 256), 256, 0, st>>>(s.agg16, s.agg, n * HID);
  gemm(st, h16, nullptr, HID, HID, s.agg16, nullptr, HID, HID, nch16, nullptr, CH_NF, CH_NF,
       P.n1, W.n1b, nullptr, s.m1, HID, n, HID, /*silu*/1, 0);
  gemm(st, s.m1, nullptr, HID, HID, nullptr, nullptr, 0, 0, nullptr, nullptr, 0, 0,
       P.n2, W.n2b, h, nullptr, HID, n, HID, 0, /*accum*/1);
  k_f32_to_f16<<<(unsigned)cdivl(n * HID, 256), 256, 0, st>>>(h16, h, n * HID);
}

extern "C" void kernel_launch(void* const* d_in, const int* in_sizes, int n_in,
                              void* d_out, int out_size, void* d_ws, size_t ws_size,
                              hipStream_t stream) {
  (void)in_sizes; (void)n_in; (void)out_size; (void)ws_size;
  hipStream_t st = stream;
  // ---- inputs ----
  const float* h_in   = (const float*)d_in[0];
  const float* x_in   = (const float*)d_in[1];
  const int*   ctxR   = (const int*)d_in[2];
  const int*   ctxC   = ctxR + E_CTX;
  const int*   imask  = (const int*)d_in[3];
  const float* ix_in  = (const float*)d_in[4];
  const int*   intR   = (const int*)d_in[5];
  const int*   intC   = intR + E_INT;
  const int*   umask  = (const int*)d_in[6];
  const int*   iumask = (const int*)d_in[7];
  const float* cha    = (const float*)d_in[8];
  const float* chw    = (const float*)d_in[9];
  const float* linin_w  = (const float*)d_in[10];
  const float* linin_b  = (const float*)d_in[11];
  const float* linout_w = (const float*)d_in[12];
  const float* linout_b = (const float*)d_in[13];
  GclW outG = gclw(d_in, 14);
  GclW ctxG[2] = { gclw(d_in, 27), gclw(d_in, 40) };
  GclW intG[2] = { gclw(d_in, 53), gclw(d_in, 66) };
  const float* gate_w = (const float*)d_in[79];
  const float* gate_b = (const float*)d_in[80];
  const float* al1_w  = (const float*)d_in[81];
  const float* al1_b  = (const float*)d_in[82];
  const float* al2_w  = (const float*)d_in[83];
  const float* al2_b  = (const float*)d_in[84];
  float* out = (float*)d_out;

  // ---- workspace bump allocator ----
  char* base = (char*)d_ws; size_t off = 0;
  auto alloc = [&](size_t bytes) -> void* {
    off = (off + 255) & ~(size_t)255;
    void* p = base + off; off += bytes; return p;
  };
  // packed weight helper (fragment layout, zeros baked in)
  auto pack = [&](const float* W, int K, int N) -> _Float16* {
    int nct = (N + 15) / 16, nch = (K + 31) / 32;
    long total = (long)nct * nch * 512;
    _Float16* p = (_Float16*)alloc((size_t)total * 2);
    k_pack_w<<<(unsigned)cdivl(total, 256), 256, 0, st>>>(W, p, K, N, nct, nch);
    return p;
  };
  auto packGcl = [&](const GclW& w) -> GclP {
    GclP p;
    p.e1 = pack(w.e1w, 2 * HID + RAD_NF, HID);
    p.e2 = pack(w.e2w, HID, HID);
    p.n1 = pack(w.n1w, 2 * HID + CH_NF, HID);
    p.n2 = pack(w.n2w, HID, HID);
    p.c1 = pack(w.c1w, HID, HID);
    p.c2 = pack(w.c2w, HID, CC);
    return p;
  };

  float*     h      = (float*)alloc((size_t)NN * HID * 4);
  _Float16*  h16    = (_Float16*)alloc((size_t)NN * HID * 2);
  _Float16*  hin16  = (_Float16*)alloc((size_t)NN * IN_NF * 2);
  float*     xb     = (float*)alloc((size_t)NN * 12 * 4);
  float*     ih     = (float*)alloc((size_t)NI_ * HID * 4);
  _Float16*  ih16   = (_Float16*)alloc((size_t)NI_ * HID * 2);
  float*     ixb    = (float*)alloc((size_t)NI_ * 12 * 4);
  float*     ichw   = (float*)alloc((size_t)NI_ * 4 * 4);
  _Float16*  nch16  = (_Float16*)alloc((size_t)NN * CH_NF * 2);
  _Float16*  nch16i = (_Float16*)alloc((size_t)NI_ * CH_NF * 2);
  GclWs ws;
  ws.cd    = (float*)alloc((size_t)E_CTX * 12 * 4);
  ws.rad16 = (_Float16*)alloc((size_t)E_CTX * RAD_NF * 2);
  ws.m1    = (_Float16*)alloc((size_t)E_CTX * HID * 2);
  ws.mm    = (_Float16*)alloc((size_t)E_CTX * HID * 2);
  ws.cw    = (float*)alloc((size_t)E_CTX * CC * 4);
  ws.agg   = (float*)alloc((size_t)NN * HID * 4);
  ws.agg16 = (_Float16*)alloc((size_t)NN * HID * 2);
  ws.aggx  = (float*)alloc((size_t)NN * 12 * 4);
  ws.cnt   = (float*)alloc((size_t)NN * 4);
  float*     ga   = (float*)alloc((size_t)NI_ * HID * 4);
  float*     aa   = (float*)alloc((size_t)NI_ * HID * 4);
  _Float16*  a116 = (_Float16*)alloc((size_t)NI_ * HID * 2);

  // ---- pack all weights to f16 fragments ----
  const _Float16* p_linin  = pack(linin_w, IN_NF, HID);
  const _Float16* p_linout = pack(linout_w, HID, OUT_NF);
  const _Float16* p_gate   = pack(gate_w, 2 * HID, HID);
  const _Float16* p_al1    = pack(al1_w, 2 * HID, HID);
  const _Float16* p_al2    = pack(al2_w, HID, HID);
  GclP outP = packGcl(outG);
  GclP ctxP[2] = { packGcl(ctxG[0]), packGcl(ctxG[1]) };
  GclP intP[2] = { packGcl(intG[0]), packGcl(intG[1]) };

  // ---- prologue ----
  k_f32_to_f16<<<(unsigned)cdivl((long)NN*IN_NF,256),256,0,st>>>(hin16, h_in, (long)NN*IN_NF);
  k_copy_f32<<<(unsigned)cdivl((long)NN*12,256),256,0,st>>>(xb, x_in, (long)NN*12);
  k_copy_f32<<<(unsigned)cdivl((long)NI_*12,256),256,0,st>>>(ixb, ix_in, (long)NI_*12);
  k_gather_rows<<<(unsigned)cdivl((long)NI_*4,256),256,0,st>>>(ichw, chw, imask, NI_, 4);
  k_node_ch<<<(unsigned)cdivl(NN,256),256,0,st>>>(cha, chw, nullptr, nch16, NN);
  k_node_ch<<<(unsigned)cdivl(NI_,256),256,0,st>>>(cha, chw, imask, nch16i, NI_);
  gemm(st, hin16, nullptr, IN_NF, IN_NF, nullptr,nullptr,0,0, nullptr,nullptr,0,0,
       p_linin, linin_b, h, h16, HID, NN, HID, 0, 0);
  k_gather_rows<<<(unsigned)cdivl((long)NI_*HID,256),256,0,st>>>(ih, h, imask, NI_, HID);

  // ---- layers ----
  for (int L = 0; L < 2; ++L) {
    run_gcl(st, h, h16, xb, ctxR, ctxC, chw, nch16, ctxG[L], ctxP[L], NN, E_CTX, ws);
    k_gather_scatter<<<(unsigned)cdivl((long)MM_*HID,256),256,0,st>>>(ih, h, iumask, umask, MM_, HID);
    k_f32_to_f16<<<(unsigned)cdivl((long)NI_*HID,256),256,0,st>>>(ih16, ih, (long)NI_*HID);
    run_gcl(st, ih, ih16, ixb, intR, intC, ichw, nch16i, intG[L], intP[L], NI_, E_INT, ws);
    // align: cat = [inter_h | h[inter_mask]]
    gemm(st, ih16, nullptr, HID, HID, h16, imask, HID, HID, nullptr,nullptr,0,0,
         p_gate, gate_b, ga, nullptr, HID, NI_, HID, /*sigmoid*/2, 0);
    gemm(st, ih16, nullptr, HID, HID, h16, imask, HID, HID, nullptr,nullptr,0,0,
         p_al1, al1_b, nullptr, a116, HID, NI_, HID, /*silu*/1, 0);
    gemm(st, a116, nullptr, HID, HID, nullptr,nullptr,0,0, nullptr,nullptr,0,0,
         p_al2, al2_b, aa, nullptr, HID, NI_, HID, 0, 0);
    k_align_combine<<<(unsigned)cdivl((long)NI_*HID,256),256,0,st>>>(ih, ga, aa, (long)NI_*HID);
    k_scatter_rows<<<(unsigned)cdivl((long)NI_*HID,256),256,0,st>>>(h, ih, imask, NI_, HID);
    k_f32_to_f16<<<(unsigned)cdivl((long)NN*HID,256),256,0,st>>>(h16, h, (long)NN*HID);
    k_f32_to_f16<<<(unsigned)cdivl((long)NI_*HID,256),256,0,st>>>(ih16, ih, (long)NI_*HID);
  }

  // ---- out GCL + lin_out ----
  run_gcl(st, h, h16, xb, ctxR, ctxC, chw, nch16, outG, outP, NN, E_CTX, ws);
  gemm(st, h16, nullptr, HID, HID, nullptr,nullptr,0,0, nullptr,nullptr,0,0,
       p_linout, linout_b, out, nullptr, OUT_NF, NN, OUT_NF, 0, 0);
  k_copy_f32<<<(unsigned)cdivl((long)NN*12,256),256,0,st>>>(out + (long)NN*OUT_NF, xb, (long)NN*12);
  k_copy_f32<<<(unsigned)cdivl((long)NI_*12,256),256,0,st>>>(out + (long)NN*OUT_NF + (long)NN*12,
                                                             ixb, (long)NI_*12);
}
